// RPN_2791728742481
// MI455X (gfx1250) — compile-verified, matrix operand order checked by reading, not run on hardware
//
#include <hip/hip_runtime.h>
#include <hip/hip_bf16.h>

// ---------------------------------------------------------------------------
// RPN head for MI455X (gfx1250, wave32, WMMA + TDM).
//  K1 pack conv weights f32 -> bf16, layout [m][k=tap*512+c]  (row-major A)
//  K2 pack base_feat f32 -> bf16 NHWC
//  K3 3x3 conv + bias + ReLU as implicit GEMM; double-buffered pipeline:
//     TDM (tensor_load_to_lds) streams the next A-tile and global B loads
//     prefetch into registers while WMMAs consume the current tile.
//  K4 pack cls/bbox 1x1 weights into 64x512 bf16 (+bias64)
//  K5 1x1 heads GEMM (M=64,N=16384,K=512); B-tile via tensor_load_to_lds
//  K6 fg softmax scores
//  K7 deterministic rank-based top-2000 + anchor decode + clip -> rois
//  K8/K9 loss partials, K10 finalize
// LDS tiles are [row][40] bf16: 64B payload + 16B pad per row so every WMMA
// fragment is exactly 2x ds_load_b128 (CDNA5 16-bit layout: k in
// {kb..kb+7} u {kb+16..kb+23}).  TDM descriptor pad_interval/pad_amount
// reproduce the same 80B row stride in hardware.
// ---------------------------------------------------------------------------

typedef __bf16 bf16;
typedef __attribute__((ext_vector_type(16))) __bf16 v16bf;
typedef __attribute__((ext_vector_type(8)))  float  v8f;
typedef __attribute__((ext_vector_type(4)))  unsigned int u32x4;
typedef __attribute__((ext_vector_type(8)))  unsigned int u32x8;

#define N_PIX   16384          // B*H*W = 4*64*64
#define KC      4608           // 9*512 reduction length for 3x3 conv
#define N_ANCH  36864          // H*W*A = 64*64*9
#define TOPK_N  2000
#define LDW     40             // LDS row stride in bf16 (80B = 64B + 16B pad)

// ---- TDM: DMA a 2-D tile (tile_m rows x tile_k bf16) into LDS -------------
// D# per CDNA5 ISA ch.8: group0 = {count, lds_addr, global_addr, type=2},
// group1 = {data_size=2B, pad_enable, pad_interval=16DW, pad_amount=4DW,
//           tensor dims/strides, tile dims}.  Issued by one wave (EXEC is
// ignored by TENSOR ops); completion via s_wait_tensorcnt.
__device__ inline void tdm_load_2d(unsigned lds_off, const bf16* gsrc,
                                   unsigned tile_k, unsigned tile_m,
                                   unsigned long long tensor_k, unsigned tensor_m) {
  const unsigned long long ga = (unsigned long long)(uintptr_t)gsrc;
  u32x4 g0;
  g0[0] = 1u;                                              // count=1, user mode
  g0[1] = lds_off;                                         // lds_addr
  g0[2] = (unsigned)ga;                                    // global_addr[31:0]
  g0[3] = ((unsigned)(ga >> 32) & 0x01FFFFFFu) | (2u << 30); // addr[56:32], type=2
  u32x8 g1;
  g1[0] = (1u << 16) | (1u << 20) | (3u << 22) | (3u << 25); // 2B, pad 64B->+16B
  g1[1] = ((unsigned)tensor_k & 0xFFFFu) << 16;            // tensor_dim0 lo
  g1[2] = ((unsigned)tensor_k >> 16) | ((tensor_m & 0xFFFFu) << 16);
  g1[3] = (tensor_m >> 16) | (tile_k << 16);               // dim1 hi, tile_dim0
  g1[4] = tile_m & 0xFFFFu;                                // tile_dim1 (tile_dim2=0)
  g1[5] = (unsigned)tensor_k;                              // dim0 stride lo
  g1[6] = (unsigned)(tensor_k >> 32);                      // dim0 stride hi
  g1[7] = 0u;
  asm volatile("tensor_load_to_lds %0, %1" :: "s"(g0), "s"(g1) : "memory");
}

// ---- WMMA fragment: 2x ds_load_b128 from an LDS tile laid out [row][LDW] --
__device__ inline v16bf frag_lds(const bf16* __restrict__ S, int row, int lane) {
  const int c  = row + (lane & 15);
  const int kb = (lane >> 4) << 3;                          // 0 or 8
  union { v16bf v; uint4 q[2]; } u;
  u.q[0] = *(const uint4*)&S[c * LDW + kb];                 // k = kb .. kb+7
  u.q[1] = *(const uint4*)&S[c * LDW + kb + 16];            // k = kb+16 .. kb+23
  return u.v;
}

// ---------------- K1: conv weights -> bf16, layout [m][k=tap*512+c] --------
__global__ void k_pack_w(const float* __restrict__ conv_w, bf16* __restrict__ w_bf) {
  const int i = blockIdx.x * 256 + threadIdx.x;   // over 512*4608
  const int m = i / KC, k = i - m * KC;
  const int r = k >> 9, c = k & 511;              // r = ky*3+kx
  w_bf[i] = (bf16)conv_w[(m * 512 + c) * 9 + r];  // OIHW source
}

// ---------------- K2: base_feat NCHW f32 -> NHWC bf16 ----------------------
__global__ void k_pack_x(const float* __restrict__ bf_in, bf16* __restrict__ x_bf) {
  const int i = blockIdx.x * 256 + threadIdx.x;   // over 16384*512
  const int p = i >> 9, c = i & 511;
  const int b = p >> 12, y = (p >> 6) & 63, x = p & 63;
  x_bf[i] = (bf16)bf_in[(((b << 9) + c) << 12) + (y << 6) + x];
}

// ---------------- K3: 3x3 conv + bias + ReLU (implicit GEMM, WMMA) ---------
__global__ __launch_bounds__(256) void k_conv3x3(
    const bf16* __restrict__ w_bf, const bf16* __restrict__ x_bf,
    const float* __restrict__ conv_b, bf16* __restrict__ y_bf) {
  __shared__ bf16 As[2][128 * LDW];   // [m][k] 128x32 (+pad), ping-pong
  __shared__ bf16 Bs[2][128 * LDW];   // [n][k] 128x32 (+pad), ping-pong

  const int tid = threadIdx.x, lane = tid & 31, wave = tid >> 5;
  const int wm = wave & 1, wn = wave >> 1;          // 2x4 wave grid -> 128x128 tile
  const int mbase = blockIdx.y * 128;
  const int nbase = blockIdx.x * 128;
  const unsigned as_off0 = (unsigned)(uintptr_t)&As[0][0];
  const unsigned as_off1 = (unsigned)(uintptr_t)&As[1][0];

  v8f acc[4][2];
  const v8f vz = {0.f, 0.f, 0.f, 0.f, 0.f, 0.f, 0.f, 0.f};
#pragma unroll
  for (int mt = 0; mt < 4; ++mt)
#pragma unroll
    for (int nt = 0; nt < 2; ++nt) acc[mt][nt] = vz;

  // B-tile cooperative-load roles: 128 px x 32 c, 32B per thread
  const int b_px = tid >> 1, b_half = tid & 1;
  const int gp = nbase + b_px;
  const int pb = gp >> 12, py = (gp >> 6) & 63, px = gp & 63;

  uint4 rb0, rb1;   // in-flight B-tile registers (next K-step)

  // im2col gather for K-step ks into rb0/rb1 (halo -> zeros)
  auto load_b_regs = [&](int ks) {
    const int r = ks >> 4;
    const int dy = r / 3 - 1, dx = r % 3 - 1;
    const int c0 = (ks & 15) << 5;
    const int yy = py + dy, xx = px + dx;
    rb0 = (uint4){0, 0, 0, 0}; rb1 = (uint4){0, 0, 0, 0};
    if (yy >= 0 && yy < 64 && xx >= 0 && xx < 64) {
      const uint4* src =
          (const uint4*)&x_bf[((((pb << 6) + yy) << 6) + xx) * 512 + c0 + (b_half << 4)];
      rb0 = src[0]; rb1 = src[1];
    }
  };

  // 8 WMMAs on one resident tile pair
  auto compute = [&](const bf16* Ac, const bf16* Bc) {
    v16bf af[4], bfr[2];
#pragma unroll
    for (int mt = 0; mt < 4; ++mt) af[mt] = frag_lds(Ac, wm * 64 + mt * 16, lane);
#pragma unroll
    for (int nt = 0; nt < 2; ++nt) bfr[nt] = frag_lds(Bc, wn * 32 + nt * 16, lane);
#pragma unroll
    for (int mt = 0; mt < 4; ++mt)
#pragma unroll
      for (int nt = 0; nt < 2; ++nt)
        acc[mt][nt] = __builtin_amdgcn_wmma_f32_16x16x32_bf16(
            false, af[mt], false, bfr[nt], (short)0, acc[mt][nt], false, false);
  };

  // one pipeline stage: fill `nxt` while computing `cur`, single barrier
  auto step = [&](const bf16* Ac, const bf16* Bc, unsigned as_nxt, bf16* Bn, int ks) {
    const bool more = (ks + 1) < 144;
    if (more) {
      if (wave == 0)   // TDM for next A-tile overlaps with this step's WMMAs
        tdm_load_2d(as_nxt, w_bf + (size_t)mbase * KC + (ks + 1) * 32, 32u, 128u,
                    (unsigned long long)KC, 512u);
      load_b_regs(ks + 1);                         // global loads in flight
    }
    compute(Ac, Bc);
    if (more) {
      uint4* dst = (uint4*)&Bn[b_px * LDW + (b_half << 4)];
      dst[0] = rb0; dst[1] = rb1;                  // ds_store_b128 x2
      if (wave == 0) __builtin_amdgcn_s_wait_tensorcnt(0);
    }
    __syncthreads();
  };

  // prologue: fill buffer 0 for ks = 0
  if (wave == 0)
    tdm_load_2d(as_off0, w_bf + (size_t)mbase * KC, 32u, 128u,
                (unsigned long long)KC, 512u);
  load_b_regs(0);
  {
    uint4* dst = (uint4*)&Bs[0][b_px * LDW + (b_half << 4)];
    dst[0] = rb0; dst[1] = rb1;
  }
  if (wave == 0) __builtin_amdgcn_s_wait_tensorcnt(0);
  __syncthreads();

  // main pipeline: K = 9 taps * 16 c-blocks = 144 steps, unrolled by 2 so the
  // ping-pong buffer selection (feeding the SGPR descriptor) is compile-time
  for (int ks = 0; ks < 144; ks += 2) {
    step(As[0], Bs[0], as_off1, Bs[1], ks);
    step(As[1], Bs[1], as_off0, Bs[0], ks + 1);
  }

  // epilogue: bias + ReLU, packed 16B stores of bf16 NHWC activation
  const int lhalf = (lane >> 4) << 3, lcol = lane & 15;
#pragma unroll
  for (int mt = 0; mt < 4; ++mt) {
    const int gm0 = mbase + wm * 64 + mt * 16 + lhalf;
#pragma unroll
    for (int nt = 0; nt < 2; ++nt) {
      const int p = nbase + wn * 32 + nt * 16 + lcol;
      union { bf16 h[8]; uint4 q; } o;
#pragma unroll
      for (int v = 0; v < 8; ++v)
        o.h[v] = (bf16)fmaxf(acc[mt][nt][v] + conv_b[gm0 + v], 0.f);
      *(uint4*)&y_bf[p * 512 + gm0] = o.q;
    }
  }
}

// ---------------- K4: pack cls(18)+bbox(36) 1x1 weights into [64][512] -----
__global__ void k_pack_head_w(const float* __restrict__ cls_w, const float* __restrict__ cls_b,
                              const float* __restrict__ bbox_w, const float* __restrict__ bbox_b,
                              bf16* __restrict__ wcb, float* __restrict__ bias64) {
  const int i = blockIdx.x * 256 + threadIdx.x;   // over 64*512
  const int m = i >> 9, k = i & 511;
  float v = 0.f;
  if (m < 18)      v = cls_w[m * 512 + k];
  else if (m < 54) v = bbox_w[(m - 18) * 512 + k];
  wcb[i] = (bf16)v;
  if (i < 64) {
    float bv = 0.f;
    if (i < 18)      bv = cls_b[i];
    else if (i < 54) bv = bbox_b[i - 18];
    bias64[i] = bv;
  }
}

// ---------------- K5: heads GEMM (M=64,N=16384,K=512) via WMMA + TDM -------
__global__ __launch_bounds__(256) void k_head(
    const bf16* __restrict__ wcb, const float* __restrict__ bias64,
    const bf16* __restrict__ y_bf, float* __restrict__ head) {
  __shared__ bf16 As[64 * LDW];       // [m][k] 64x32
  __shared__ bf16 Bs[128 * LDW];      // [n][k] 128x32
  const int tid = threadIdx.x, lane = tid & 31, wave = tid >> 5;
  const int mt = wave & 3, wn = wave >> 2;          // 4x2 waves -> 64x128 tile
  const int nbase = blockIdx.x * 128;
  const unsigned bs_off = (unsigned)(uintptr_t)&Bs[0];

  v8f acc[4];
  const v8f vz = {0.f, 0.f, 0.f, 0.f, 0.f, 0.f, 0.f, 0.f};
#pragma unroll
  for (int nt = 0; nt < 4; ++nt) acc[nt] = vz;

  const int a_m = tid >> 2, a_seg = (tid & 3) << 3;  // A: 64 rows x 32, 16B/thread

  for (int kk = 0; kk < 16; ++kk) {
    // B tile: TDM DMAs 128 pixels x 32 channels from the NHWC activation
    if (wave == 0)
      tdm_load_2d(bs_off, y_bf + (size_t)nbase * 512 + (kk << 5), 32u, 128u,
                  512ull, (unsigned)N_PIX);
    { // A tile: manual 16B copies
      const uint4* src = (const uint4*)&wcb[a_m * 512 + (kk << 5) + a_seg];
      *(uint4*)&As[a_m * LDW + a_seg] = src[0];
    }
    if (wave == 0) __builtin_amdgcn_s_wait_tensorcnt(0);
    __syncthreads();

    const v16bf af = frag_lds(As, mt * 16, lane);
#pragma unroll
    for (int nt = 0; nt < 4; ++nt) {
      const v16bf bfr = frag_lds(Bs, wn * 64 + nt * 16, lane);
      acc[nt] = __builtin_amdgcn_wmma_f32_16x16x32_bf16(
          false, af, false, bfr, (short)0, acc[nt], false, false);
    }
    __syncthreads();
  }

  const int lhalf = (lane >> 4) << 3, lcol = lane & 15;
  const int gm0 = mt * 16 + lhalf;
#pragma unroll
  for (int nt = 0; nt < 4; ++nt) {
    const int p = nbase + wn * 64 + nt * 16 + lcol;
#pragma unroll
    for (int v = 0; v < 8; ++v)
      head[(gm0 + v) * N_PIX + p] = acc[nt][v] + bias64[gm0 + v];
  }
}

// ---------------- K6: foreground softmax scores ----------------------------
__global__ void k_fg(const float* __restrict__ head, float* __restrict__ fs) {
  const int e = blockIdx.x * 256 + threadIdx.x;     // over B*N_ANCH = 147456
  const int b = e / N_ANCH;
  const int n = e - b * N_ANCH;
  const int a = n % 9, pix = n / 9;
  const int idx = (b << 12) + pix;
  const float sb = head[a * N_PIX + idx];
  const float sf = head[(9 + a) * N_PIX + idx];
  fs[e] = 1.f / (1.f + expf(sb - sf));
}

// ---------------- K7: deterministic rank top-k + decode + clip -------------
__global__ __launch_bounds__(256) void k_topk(
    const float* __restrict__ fs, const float* __restrict__ head,
    const float* __restrict__ im_info, float* __restrict__ rois) {
  const int b = blockIdx.y;
  const int e = blockIdx.x * 256 + threadIdx.x;     // e < 36864
  const float* s = fs + b * N_ANCH;
  const float myv = s[e];
  int rank = 0;
  __shared__ float tile[256];
  for (int jt = 0; jt < 144; ++jt) {
    __syncthreads();
    tile[threadIdx.x] = s[(jt << 8) + threadIdx.x];
    __syncthreads();
    const int jbase = jt << 8;
#pragma unroll 8
    for (int j = 0; j < 256; ++j) {
      const float v = tile[j];
      if (v > myv || (v == myv && (jbase + j) < e)) ++rank;   // top_k tie order
    }
  }
  if (rank < TOPK_N) {
    // anchor tables: a = ratio*3 + scale; widths/heights of base-16 anchors
    const float AW[9] = {184.f, 368.f, 736.f, 128.f, 256.f, 512.f, 88.f, 176.f, 352.f};
    const float AH[9] = {96.f, 192.f, 384.f, 128.f, 256.f, 512.f, 176.f, 352.f, 704.f};
    const int a = e % 9, pix = e / 9;
    const int y = pix >> 6, x = pix & 63;
    const int idx = (b << 12) + pix;
    const float d0 = head[(18 + 4 * a + 0) * N_PIX + idx];
    const float d1 = head[(18 + 4 * a + 1) * N_PIX + idx];
    const float d2 = head[(18 + 4 * a + 2) * N_PIX + idx];
    const float d3 = head[(18 + 4 * a + 3) * N_PIX + idx];
    const float wa = AW[a], ha = AH[a];
    const float cx = d0 * wa + (8.f + 16.f * (float)x);
    const float cy = d1 * ha + (8.f + 16.f * (float)y);
    const float pw = expf(d2) * wa, ph = expf(d3) * ha;
    const float limw = im_info[b * 3 + 1] - 1.f;
    const float limh = im_info[b * 3 + 0] - 1.f;
    float* dst = rois + (size_t)(b * TOPK_N + rank) * 5;
    dst[0] = (float)b;
    dst[1] = fminf(fmaxf(cx - 0.5f * pw, 0.f), limw);
    dst[2] = fminf(fmaxf(cy - 0.5f * ph, 0.f), limh);
    dst[3] = fminf(fmaxf(cx + 0.5f * pw - 1.f, 0.f), limw);
    dst[4] = fminf(fmaxf(cy + 0.5f * ph - 1.f, 0.f), limh);
  }
}

// ---------------- fixed-order block reduction ------------------------------
__device__ inline float blk_reduce(float v, float* sm) {
  const int t = threadIdx.x;
  __syncthreads();
  sm[t] = v; __syncthreads();
  for (int s = 128; s > 0; s >>= 1) { if (t < s) sm[t] += sm[t + s]; __syncthreads(); }
  const float r = sm[0];
  __syncthreads();
  return r;
}

// ---------------- K8: CE-on-probs loss partials ----------------------------
__global__ __launch_bounds__(256) void k_loss_cls(
    const float* __restrict__ head, const int* __restrict__ lab,
    float* __restrict__ psum, float* __restrict__ pcnt) {
  const int e = blockIdx.x * 256 + threadIdx.x;     // over 147456, [b][a][h][w]
  float s = 0.f, c = 0.f;
  const int l = lab[e];
  if (l != -1) {
    const int b = e / N_ANCH;
    const int rem = e - b * N_ANCH;
    const int a = rem >> 12, pix = rem & 4095;
    const int idx = (b << 12) + pix;
    const float sb = head[a * N_PIX + idx];
    const float sf = head[(9 + a) * N_PIX + idx];
    const float qf = 1.f / (1.f + expf(sb - sf));   // softmax prob (prob2)
    const float qb = 1.f - qf;
    const float mx = fmaxf(qb, qf);                 // log_softmax over the *probs*
    const float lse = mx + logf(expf(qb - mx) + expf(qf - mx));
    s = lse - (l == 1 ? qf : qb);
    c = 1.f;
  }
  __shared__ float sm[256];
  const float ts = blk_reduce(s, sm);
  const float tc = blk_reduce(c, sm);
  if (threadIdx.x == 0) { psum[blockIdx.x] = ts; pcnt[blockIdx.x] = tc; }
}

// ---------------- K9: smooth-L1 box loss partials --------------------------
__global__ __launch_bounds__(256) void k_loss_box(
    const float* __restrict__ head, const float* __restrict__ tgt,
    const float* __restrict__ iw, const float* __restrict__ ow,
    float* __restrict__ pbox) {
  const int e = blockIdx.x * 256 + threadIdx.x;     // over 589824, NCHW [b][36][h][w]
  const int b = e / 147456;
  const int rem = e - b * 147456;
  const int c = rem >> 12, pix = rem & 4095;
  const float off = head[(18 + c) * N_PIX + (b << 12) + pix];
  const float d = iw[e] * (off - tgt[e]);
  const float ad = fabsf(d);
  const float l = (ad < (1.f / 9.f)) ? d * d * 4.5f : (ad - 0.5f / 9.f);
  __shared__ float sm[256];
  const float ts = blk_reduce(ow[e] * l, sm);
  if (threadIdx.x == 0) pbox[blockIdx.x] = ts;
}

// ---------------- K10: finalize losses -------------------------------------
__global__ __launch_bounds__(256) void k_finalize(
    const float* __restrict__ psum, const float* __restrict__ pcnt,
    const float* __restrict__ pbox, float* __restrict__ out_losses) {
  float s = 0.f, c = 0.f, bx = 0.f;
  for (int i = threadIdx.x; i < 576; i += 256) { s += psum[i]; c += pcnt[i]; }
  for (int i = threadIdx.x; i < 2304; i += 256) bx += pbox[i];
  __shared__ float sm[256];
  s = blk_reduce(s, sm); c = blk_reduce(c, sm); bx = blk_reduce(bx, sm);
  if (threadIdx.x == 0) {
    out_losses[0] = s / fmaxf(c, 1.f);
    out_losses[1] = bx * 0.25f;      // mean over B=4 of per-batch sums
  }
}

// ---------------------------------------------------------------------------
extern "C" void kernel_launch(void* const* d_in, const int* in_sizes, int n_in,
                              void* d_out, int out_size, void* d_ws, size_t ws_size,
                              hipStream_t stream) {
  (void)in_sizes; (void)n_in; (void)out_size; (void)ws_size;
  const float* base_feat = (const float*)d_in[0];
  const float* im_info   = (const float*)d_in[1];
  const float* conv_w    = (const float*)d_in[2];
  const float* conv_b    = (const float*)d_in[3];
  const float* cls_w     = (const float*)d_in[4];
  const float* cls_b     = (const float*)d_in[5];
  const float* bbox_w    = (const float*)d_in[6];
  const float* bbox_b    = (const float*)d_in[7];
  const int*   rpn_label = (const int*)d_in[8];
  const float* bbox_t    = (const float*)d_in[9];
  const float* bbox_iw   = (const float*)d_in[10];
  const float* bbox_ow   = (const float*)d_in[11];
  float* out = (float*)d_out;           // [B*2000*5 rois][loss_cls][loss_box]

  // workspace carve-out (~43 MB)
  char* ws = (char*)d_ws;
  size_t o = 0;
  auto carve = [&](size_t bytes) -> void* {
    void* p = ws + o;
    o += (bytes + 255) & ~(size_t)255;
    return p;
  };
  bf16*  w_bf   = (bf16*) carve((size_t)512 * KC * 2);       // 4.7 MB  [m][k]
  bf16*  x_bf   = (bf16*) carve((size_t)N_PIX * 512 * 2);    // 16.8 MB NHWC
  bf16*  y_bf   = (bf16*) carve((size_t)N_PIX * 512 * 2);    // 16.8 MB NHWC
  bf16*  wcb    = (bf16*) carve((size_t)64 * 512 * 2);       // [m][k]
  float* bias64 = (float*)carve(64 * 4);
  float* headb  = (float*)carve((size_t)64 * N_PIX * 4);     // 4.2 MB
  float* fs     = (float*)carve((size_t)4 * N_ANCH * 4);     // 0.6 MB
  float* psum   = (float*)carve(576 * 4);
  float* pcnt   = (float*)carve(576 * 4);
  float* pbox   = (float*)carve(2304 * 4);

  k_pack_w     <<<9216,  256, 0, stream>>>(conv_w, w_bf);
  k_pack_x     <<<32768, 256, 0, stream>>>(base_feat, x_bf);
  k_pack_head_w<<<128,   256, 0, stream>>>(cls_w, cls_b, bbox_w, bbox_b, wcb, bias64);
  k_conv3x3    <<<dim3(128, 4), 256, 0, stream>>>(w_bf, x_bf, conv_b, y_bf);
  k_head       <<<128,   256, 0, stream>>>(wcb, bias64, y_bf, headb);
  k_fg         <<<576,   256, 0, stream>>>(headb, fs);
  k_topk       <<<dim3(144, 4), 256, 0, stream>>>(fs, headb, im_info, out);
  k_loss_cls   <<<576,   256, 0, stream>>>(headb, rpn_label, psum, pcnt);
  k_loss_box   <<<2304,  256, 0, stream>>>(headb, bbox_t, bbox_iw, bbox_ow, pbox);
  k_finalize   <<<1,     256, 0, stream>>>(psum, pcnt, pbox, out + 4 * TOPK_N * 5);
}
